// GlobalSelfAttentionV3_2559800509205
// MI455X (gfx1250) — matrix operand order repaired and checked by reading, not compiled, not run on hardware
//
#include <hip/hip_runtime.h>
#include <hip/hip_bf16.h>
#include <math.h>

typedef __attribute__((ext_vector_type(16))) _Float16 v16h;
typedef __attribute__((ext_vector_type(8)))  float    v8f;

#define BATCH 32
#define DIM   512
#define C     128
#define WS    14
#define NH    4
#define HD    32
#define NTOK  196
#define NP    224      // padded tokens: 14*16 = 7*32
#define HIN   56

// ---------------------------------------------------------------------------
// K1: grouped 4x4/s4 conv + BN(eval) + ReLU6 -> f16 tokens [B][NP][C] (pad=0)
// ---------------------------------------------------------------------------
__global__ void __launch_bounds__(256)
k_conv_tokens(const float* __restrict__ x, const float* __restrict__ conv_w,
              const float* __restrict__ conv_b, const float* __restrict__ bn_g,
              const float* __restrict__ bn_b, const float* __restrict__ bn_m,
              const float* __restrict__ bn_v, _Float16* __restrict__ t_h) {
  int idx = blockIdx.x * 256 + threadIdx.x;              // B*NP*C
  if (idx >= BATCH * NP * C) return;
  int c   = idx & (C - 1);
  int tok = (idx >> 7) % NP;
  int b   = idx / (NP * C);
  if (tok >= NTOK) { t_h[idx] = (_Float16)0.0f; return; }
  int oy = tok / WS, ox = tok % WS;
  float acc = conv_b[c];
  const float* xb = x + ((size_t)b * DIM + (size_t)c * 4) * (HIN * HIN);
  const float* w  = conv_w + c * 64;
#pragma unroll
  for (int i = 0; i < 4; ++i)
#pragma unroll
    for (int ky = 0; ky < 4; ++ky)
#pragma unroll
      for (int kx = 0; kx < 4; ++kx)
        acc += xb[(size_t)i * (HIN * HIN) + (oy * 4 + ky) * HIN + (ox * 4 + kx)]
               * w[i * 16 + ky * 4 + kx];
  float s = bn_g[c] * rsqrtf(bn_v[c] + 1e-5f);
  float y = acc * s + (bn_b[c] - bn_m[c] * s);
  y = fminf(fmaxf(y, 0.0f), 6.0f);
  t_h[idx] = (_Float16)y;
}

// ---------------------------------------------------------------------------
// K2: fold proj into convout: W2[od][in] = sum_j convout_w[od][j]*proj_w[j][in]
//     fb[od] = convout_w[od]·proj_b + convout_b[od]
// ---------------------------------------------------------------------------
__global__ void __launch_bounds__(256)
k_fuse_w(const float* __restrict__ proj_w, const float* __restrict__ proj_b,
         const float* __restrict__ convout_w, const float* __restrict__ convout_b,
         _Float16* __restrict__ W2, float* __restrict__ fb) {
  int idx = blockIdx.x * 256 + threadIdx.x;              // DIM*C
  if (idx >= DIM * C) return;
  int in = idx & (C - 1);
  int od = idx >> 7;
  float acc = 0.0f;
  for (int j = 0; j < C; ++j)
    acc += convout_w[od * C + j] * proj_w[j * C + in];
  W2[idx] = (_Float16)acc;
  if (in == 0) {
    float a = convout_b[od];
    for (int j = 0; j < C; ++j) a += convout_w[od * C + j] * proj_b[j];
    fb[od] = a;
  }
}

// ---------------------------------------------------------------------------
// K3: QKV GEMM: [NP x C] x [C x 3C] -> q/k [B][NH][NP][HD] (q pre-scaled),
//     v stored TRANSPOSED: vT [B][NH][HD][NP] so PV B-fragments are contiguous.
// grid (B, 14, 3), block 256 (8 waves -> 8 n-tiles = 128 cols)
// ---------------------------------------------------------------------------
__global__ void __launch_bounds__(256)
k_qkv(const _Float16* __restrict__ t_h, const float* __restrict__ qkv_w,
      _Float16* __restrict__ q_h, _Float16* __restrict__ k_h,
      _Float16* __restrict__ vT_h) {
  __shared__ _Float16 lA[16 * 128];
  __shared__ _Float16 lB[128 * 128];
  int b = blockIdx.x, m0 = blockIdx.y * 16, n0 = blockIdx.z * 128;
  int tid = threadIdx.x;
  const _Float16* tb = t_h + ((size_t)b * NP + m0) * C;
  for (int i = tid; i < 16 * 128; i += 256) lA[i] = tb[i];
  for (int i = tid; i < 128 * 128; i += 256)
    lB[i] = (_Float16)qkv_w[(size_t)(n0 + (i >> 7)) * C + (i & 127)];
  __syncthreads();

  int wave = tid >> 5, lane = tid & 31, lo = lane & 15, hi = lane >> 4;
  v8f acc = {};
  for (int k0 = 0; k0 < 128; k0 += 32) {
    v16h a, bm;
#pragma unroll
    for (int e = 0; e < 16; ++e) {
      a[e]  = lA[lo * 128 + k0 + 16 * (e >> 3) + 8 * hi + (e & 7)];
      bm[e] = lB[(wave * 16 + lo) * 128 + k0 + 16 * (e >> 3) + 8 * hi + (e & 7)];
    }
    acc = __builtin_amdgcn_wmma_f32_16x16x32_f16(false, a, false, bm,
                                                 (short)0, acc, false, false);
  }
  int ncol = n0 + wave * 16 + lo;
  int which = ncol >> 7;              // uniform per block (depends on n0 only)
  int head = (ncol >> 5) & 3, hd = ncol & 31;
  if (which == 2) {
    // vT[b][head][hd][tok]: 8 consecutive tokens per lane -> contiguous store
    _Float16* dst = vT_h + (((size_t)b * NH + head) * HD + hd) * NP + m0 + 8 * hi;
#pragma unroll
    for (int r = 0; r < 8; ++r) dst[r] = (_Float16)acc[r];
  } else {
    _Float16* dst = (which == 0) ? q_h : k_h;
    float sc = (which == 0) ? 0.17677669529663687f : 1.0f;   // HD^-0.5
#pragma unroll
    for (int r = 0; r < 8; ++r) {
      int tok = m0 + r + 8 * hi;
      dst[(((size_t)b * NH + head) * NP + tok) * HD + hd] = (_Float16)(acc[r] * sc);
    }
  }
}

// ---------------------------------------------------------------------------
// K4: scores = q·k^T + rel-pos bias.  grid (B*NH, 14, 14), block 32 (1 wave)
// ---------------------------------------------------------------------------
__global__ void __launch_bounds__(32)
k_scores(const _Float16* __restrict__ q_h, const _Float16* __restrict__ k_h,
         const float* __restrict__ rpb, float* __restrict__ scores) {
  int bh = blockIdx.x;
  int h  = bh & 3;
  int m0 = blockIdx.y * 16, n0 = blockIdx.z * 16;
  int lane = threadIdx.x, lo = lane & 15, hi = lane >> 4;
  const _Float16* qb = q_h + (size_t)bh * NP * HD;
  const _Float16* kb = k_h + (size_t)bh * NP * HD;
  v16h a, bm;
#pragma unroll
  for (int e = 0; e < 16; ++e) {
    a[e]  = qb[(m0 + lo) * HD + 16 * (e >> 3) + 8 * hi + (e & 7)];
    bm[e] = kb[(n0 + lo) * HD + 16 * (e >> 3) + 8 * hi + (e & 7)];
  }
  v8f acc = {};
  acc = __builtin_amdgcn_wmma_f32_16x16x32_f16(false, a, false, bm,
                                               (short)0, acc, false, false);
  int ncol = n0 + lo;
  int yn = ncol / WS, xn = ncol % WS;
#pragma unroll
  for (int r = 0; r < 8; ++r) {
    int m = m0 + r + 8 * hi;
    float s;
    if (ncol < NTOK && m < NTOK) {
      int ym = m / WS, xm = m % WS;
      int idx = (ym - yn + WS - 1) * (2 * WS - 1) + (xm - xn + WS - 1);
      s = acc[r] + rpb[idx * NH + h];
    } else {
      s = -1e30f;                 // pad -> exp() == 0 after softmax
    }
    scores[((size_t)bh * NP + m) * NP + ncol] = s;
  }
}

// ---------------------------------------------------------------------------
// K5: row softmax -> P (f16).  one wave per row, 8 rows per block
// ---------------------------------------------------------------------------
__global__ void __launch_bounds__(256)
k_softmax(const float* __restrict__ scores, _Float16* __restrict__ p) {
  int row  = blockIdx.x * 8 + (threadIdx.x >> 5);          // B*NH*NP rows
  int lane = threadIdx.x & 31;
  const float* src = scores + (size_t)row * NP;
  float v[7];
  float mx = -3.0e30f;
#pragma unroll
  for (int i = 0; i < 7; ++i) { v[i] = src[i * 32 + lane]; mx = fmaxf(mx, v[i]); }
#pragma unroll
  for (int off = 16; off; off >>= 1) mx = fmaxf(mx, __shfl_xor(mx, off, 32));
  float sum = 0.0f;
#pragma unroll
  for (int i = 0; i < 7; ++i) { v[i] = __expf(v[i] - mx); sum += v[i]; }
#pragma unroll
  for (int off = 16; off; off >>= 1) sum += __shfl_xor(sum, off, 32);
  float inv = 1.0f / sum;
  _Float16* dst = p + (size_t)row * NP;
#pragma unroll
  for (int i = 0; i < 7; ++i) dst[i * 32 + lane] = (_Float16)(v[i] * inv);
}

// ---------------------------------------------------------------------------
// K6: o = P·V (V transposed).  grid (B*NH, 14), block 64 (2 waves), K=224
// B-fragment: vT[col_hd][k] contiguous in k -> vectorized b128 loads
// ---------------------------------------------------------------------------
__global__ void __launch_bounds__(64)
k_av(const _Float16* __restrict__ p, const _Float16* __restrict__ vT_h,
     _Float16* __restrict__ o_h) {
  int bh = blockIdx.x;
  int b = bh >> 2, h = bh & 3;
  int m0 = blockIdx.y * 16;
  int wave = threadIdx.x >> 5, lane = threadIdx.x & 31, lo = lane & 15, hi = lane >> 4;
  int n0 = wave * 16;
  const _Float16* pb = p + (size_t)bh * NP * NP;
  const _Float16* vb = vT_h + (size_t)bh * HD * NP;
  v8f acc = {};
  for (int k0 = 0; k0 < NP; k0 += 32) {
    v16h a, bm;
#pragma unroll
    for (int e = 0; e < 16; ++e) {
      a[e]  = pb[(m0 + lo) * NP + k0 + 16 * (e >> 3) + 8 * hi + (e & 7)];
      bm[e] = vb[(n0 + lo) * NP + k0 + 16 * (e >> 3) + 8 * hi + (e & 7)];
    }
    acc = __builtin_amdgcn_wmma_f32_16x16x32_f16(false, a, false, bm,
                                                 (short)0, acc, false, false);
  }
#pragma unroll
  for (int r = 0; r < 8; ++r) {
    int tok = m0 + r + 8 * hi;
    o_h[((size_t)b * NP + tok) * C + h * HD + n0 + lo] = (_Float16)acc[r];
  }
}

// ---------------------------------------------------------------------------
// K7: feat = o·W2^T + fb -> [B][512][196].  grid (B, 14, 4), block 256
// ---------------------------------------------------------------------------
__global__ void __launch_bounds__(256)
k_out(const _Float16* __restrict__ o_h, const _Float16* __restrict__ W2,
      const float* __restrict__ fb, float* __restrict__ feat) {
  __shared__ _Float16 lA[16 * 128];
  __shared__ _Float16 lB[128 * 128];
  int b = blockIdx.x, m0 = blockIdx.y * 16, n0 = blockIdx.z * 128;
  int tid = threadIdx.x;
  const _Float16* ob = o_h + ((size_t)b * NP + m0) * C;
  for (int i = tid; i < 16 * 128; i += 256) lA[i] = ob[i];
  for (int i = tid; i < 128 * 128; i += 256)
    lB[i] = W2[(size_t)(n0 + (i >> 7)) * C + (i & 127)];
  __syncthreads();

  int wave = tid >> 5, lane = tid & 31, lo = lane & 15, hi = lane >> 4;
  v8f acc = {};
  for (int k0 = 0; k0 < 128; k0 += 32) {
    v16h a, bm;
#pragma unroll
    for (int e = 0; e < 16; ++e) {
      a[e]  = lA[lo * 128 + k0 + 16 * (e >> 3) + 8 * hi + (e & 7)];
      bm[e] = lB[(wave * 16 + lo) * 128 + k0 + 16 * (e >> 3) + 8 * hi + (e & 7)];
    }
    acc = __builtin_amdgcn_wmma_f32_16x16x32_f16(false, a, false, bm,
                                                 (short)0, acc, false, false);
  }
  int od = n0 + wave * 16 + lo;
  float bias = fb[od];
#pragma unroll
  for (int r = 0; r < 8; ++r) {
    int tok = m0 + r + 8 * hi;
    if (tok < NTOK)
      feat[((size_t)b * DIM + od) * NTOK + tok] = acc[r] + bias;
  }
}

// ---------------------------------------------------------------------------
// K8: bilinear x4 upsample (align_corners=True): [B,512,14,14] -> [B,512,56,56]
// 4 consecutive output pixels per thread -> float4 (b128) stores
// ---------------------------------------------------------------------------
__global__ void __launch_bounds__(256)
k_upsample(const float* __restrict__ feat, float* __restrict__ out) {
  size_t idx = (size_t)blockIdx.x * 256 + threadIdx.x;    // B*DIM*56*14 quads
  if (idx >= (size_t)BATCH * DIM * HIN * (HIN / 4)) return;
  int quad = (int)(idx % (HIN / 4));
  int O    = (int)((idx / (HIN / 4)) % HIN);
  int d    = (int)((idx / ((HIN / 4) * HIN)) % DIM);
  int b    = (int)(idx / ((size_t)(HIN / 4) * HIN * DIM));

  float sy = O * (13.0f / 55.0f);
  int y0 = (int)sy;
  float wy = sy - y0;
  int y1 = (y0 + 1 < WS) ? y0 + 1 : WS - 1;
  const float* f0 = feat + ((size_t)b * DIM + d) * NTOK + y0 * WS;
  const float* f1 = feat + ((size_t)b * DIM + d) * NTOK + y1 * WS;

  float4 res;
  float r[4];
#pragma unroll
  for (int j = 0; j < 4; ++j) {
    int P = quad * 4 + j;
    float sx = P * (13.0f / 55.0f);
    int x0 = (int)sx;
    float wx = sx - x0;
    int x1 = (x0 + 1 < WS) ? x0 + 1 : WS - 1;
    float top = (1.0f - wx) * f0[x0] + wx * f0[x1];
    float bot = (1.0f - wx) * f1[x0] + wx * f1[x1];
    r[j] = (1.0f - wy) * top + wy * bot;
  }
  res.x = r[0]; res.y = r[1]; res.z = r[2]; res.w = r[3];
  float4* dst = (float4*)(out + (((size_t)b * DIM + d) * HIN + O) * HIN + quad * 4);
  *dst = res;
}

// ---------------------------------------------------------------------------
extern "C" void kernel_launch(void* const* d_in, const int* in_sizes, int n_in,
                              void* d_out, int out_size, void* d_ws, size_t ws_size,
                              hipStream_t stream) {
  const float* x         = (const float*)d_in[0];
  const float* conv_w    = (const float*)d_in[1];
  const float* conv_b    = (const float*)d_in[2];
  const float* bn_gamma  = (const float*)d_in[3];
  const float* bn_beta   = (const float*)d_in[4];
  const float* bn_mean   = (const float*)d_in[5];
  const float* bn_var    = (const float*)d_in[6];
  const float* qkv_w     = (const float*)d_in[7];
  const float* proj_w    = (const float*)d_in[8];
  const float* proj_b    = (const float*)d_in[9];
  const float* rpb_table = (const float*)d_in[10];
  const float* convout_w = (const float*)d_in[11];
  const float* convout_b = (const float*)d_in[12];
  float* out = (float*)d_out;

  // ---- workspace carve (256B aligned) ----
  char* w = (char*)d_ws;
  auto carve = [&](size_t bytes) {
    char* p = w;
    w += (bytes + 255) & ~(size_t)255;
    return p;
  };
  _Float16* t_h    = (_Float16*)carve((size_t)BATCH * NP * C * 2);
  _Float16* q_h    = (_Float16*)carve((size_t)BATCH * NH * NP * HD * 2);
  _Float16* k_h    = (_Float16*)carve((size_t)BATCH * NH * NP * HD * 2);
  _Float16* vT_h   = (_Float16*)carve((size_t)BATCH * NH * HD * NP * 2);
  float*    scores = (float*)   carve((size_t)BATCH * NH * NP * NP * 4);
  _Float16* p_h    = (_Float16*)carve((size_t)BATCH * NH * NP * NP * 2);
  _Float16* o_h    = (_Float16*)carve((size_t)BATCH * NP * C * 2);
  _Float16* W2     = (_Float16*)carve((size_t)DIM * C * 2);
  float*    fb     = (float*)   carve((size_t)DIM * 4);
  float*    feat   = (float*)   carve((size_t)BATCH * DIM * NTOK * 4);
  (void)ws_size; (void)n_in; (void)in_sizes; (void)out_size;

  // K2 (independent) + K1
  k_fuse_w<<<(DIM * C + 255) / 256, 256, 0, stream>>>(proj_w, proj_b,
                                                      convout_w, convout_b, W2, fb);
  k_conv_tokens<<<(BATCH * NP * C + 255) / 256, 256, 0, stream>>>(
      x, conv_w, conv_b, bn_gamma, bn_beta, bn_mean, bn_var, t_h);

  // K3: qkv
  {
    dim3 g(BATCH, NP / 16, 3);
    k_qkv<<<g, 256, 0, stream>>>(t_h, qkv_w, q_h, k_h, vT_h);
  }
  // K4: scores + bias
  {
    dim3 g(BATCH * NH, NP / 16, NP / 16);
    k_scores<<<g, 32, 0, stream>>>(q_h, k_h, rpb_table, scores);
  }
  // K5: softmax
  k_softmax<<<(BATCH * NH * NP) / 8, 256, 0, stream>>>(scores, p_h);
  // K6: P·V
  {
    dim3 g(BATCH * NH, NP / 16);
    k_av<<<g, 64, 0, stream>>>(p_h, vT_h, o_h);
  }
  // K7: fused proj + 1x1 convout
  {
    dim3 g(BATCH, NP / 16, 4);
    k_out<<<g, 256, 0, stream>>>(o_h, W2, fb, feat);
  }
  // K8: upsample (float4 stores)
  {
    size_t total = (size_t)BATCH * DIM * HIN * (HIN / 4);
    k_upsample<<<(unsigned)((total + 255) / 256), 256, 0, stream>>>(feat, out);
  }
}
